// STGAT_61658550502047
// MI455X (gfx1250) — matrix operand order repaired and checked by reading, not compiled, not run on hardware
//
#include <hip/hip_runtime.h>
#include <math.h>

typedef __attribute__((ext_vector_type(16))) _Float16 v16h;
typedef __attribute__((ext_vector_type(8)))  float    v8f;

#define B_     4
#define T_     12
#define N_     5000
#define FIN_   32
#define HID_   32
#define HEADS_ 4
#define OUT_   12
#define E_     40000
#define D1_    128
#define ESL_   (E_ + N_)      // edges + self loops
#define G_     (B_ * T_)      // 48 graph snapshots

// ======================= WMMA GEMM ==========================================
// One wave computes one 16x16 f32 tile:  D = A1@B1 (+ A2@B2) (+bias) (+relu).
// A: f32 in global, converted to f16 on load in contiguous k order (k = 16*hi + e)
//    (compiler vectorizes the contiguous 16-float run into b128 loads).
// B: pre-packed into fragment order: per (nTile,kTile) fragment, 32 lanes x
//    16 contiguous halves -> one aligned v16h (2x global_load_b128) per lane.
__device__ __forceinline__ v8f wmma_k32(const float* __restrict__ ap,      // &A[r0*lda + kt*32]
                                        const _Float16* __restrict__ bfr,  // lane's packed 16 halves
                                        int hi, v8f acc)
{
    v16h af;
#pragma unroll
    for (int e = 0; e < 16; ++e) {
        // lanes 0-15: K = e;  lanes 16-31: K = 16+e  (matches k_pack B order)
        int k = e + hi * 16;
        af[e] = (_Float16)ap[k];
    }
    v16h bf = *(const v16h*)bfr;
    return __builtin_amdgcn_wmma_f32_16x16x32_f16(false, af, false, bf,
                                                  (short)0, acc, false, false);
}

__global__ void __launch_bounds__(256)
k_gemm(const float* __restrict__ A1, size_t sA1, int lda1, int K1,
       const float* __restrict__ A2, size_t sA2, int lda2, int K2,
       const _Float16* __restrict__ B1p, const _Float16* __restrict__ B2p,
       const float* __restrict__ bias, int relu,
       float* __restrict__ Out, size_t sOut, int ldo,
       int M, int Ncols, int NStore)
{
    const int lane  = threadIdx.x & 31;
    const int wave  = threadIdx.x >> 5;
    const int nTN   = Ncols >> 4;
    const int nTM   = (M + 15) >> 4;
    const int tile  = blockIdx.x * 8 + wave;
    if (tile >= nTM * nTN) return;            // wave-uniform: EXEC stays all-ones
    const int mT    = tile / nTN;
    const int nT    = tile - mT * nTN;
    const int mBase = mT << 4, nBase = nT << 4;
    const int l16   = lane & 15, hi = lane >> 4;
    int r0 = mBase + l16;                      // A row for this lane
    if (r0 >= M) r0 = M - 1;                   // clamp (store is predicated)

    const int nKt1 = K1 >> 5;
    const float* a1 = A1 + (size_t)blockIdx.y * sA1 + (size_t)r0 * lda1;
    const _Float16* b1 = B1p + ((size_t)nT * nKt1) * 512 + (size_t)lane * 16;
    v8f acc = {};
    for (int kt = 0; kt < nKt1; ++kt)
        acc = wmma_k32(a1 + kt * 32, b1 + (size_t)kt * 512, hi, acc);
    if (A2) {
        const int nKt2 = K2 >> 5;
        const float* a2 = A2 + (size_t)blockIdx.y * sA2 + (size_t)r0 * lda2;
        const _Float16* b2 = B2p + ((size_t)nT * nKt2) * 512 + (size_t)lane * 16;
        for (int kt = 0; kt < nKt2; ++kt)
            acc = wmma_k32(a2 + kt * 32, b2 + (size_t)kt * 512, hi, acc);
    }
    const int  col   = nBase + l16;
    const bool colOK = col < NStore;
    float bv = (bias && colOK) ? bias[col] : 0.f;
    float* outp = Out + (size_t)blockIdx.y * sOut;
#pragma unroll
    for (int r = 0; r < 8; ++r) {              // D layout: VGPR r -> row r + hi*8
        int row = mBase + hi * 8 + r;
        if (colOK && row < M) {
            float v = acc[r] + bv;
            if (relu) v = fmaxf(v, 0.f);
            outp[(size_t)row * ldo + col] = v;
        }
    }
}

// ======================= weight prep ========================================
__global__ void k_f2h(const float* __restrict__ s, _Float16* __restrict__ d, int n) {
    int i = blockIdx.x * 256 + threadIdx.x;
    if (i < n) d[i] = (_Float16)s[i];
}
// dst[k*96 + g*32 + c] = Wg[order*K*32 + k*32 + c]   (gate concat z|r|h)
__global__ void k_cat3(const float* __restrict__ wz, const float* __restrict__ wr,
                       const float* __restrict__ wh, _Float16* __restrict__ dst,
                       int K, int order) {
    int i = blockIdx.x * 256 + threadIdx.x;
    if (i >= K * 96) return;
    int k = i / 96, j = i - k * 96, g = j >> 5, c = j & 31;
    const float* w = (g == 0) ? wz : (g == 1) ? wr : wh;
    dst[i] = (_Float16)w[(size_t)order * K * 32 + (size_t)k * 32 + c];
}
__global__ void k_cat2(const float* __restrict__ w0, const float* __restrict__ w1,
                       _Float16* __restrict__ dst, int K, int order) {
    int i = blockIdx.x * 256 + threadIdx.x;
    if (i >= K * 64) return;
    int k = i / 64, j = i - k * 64, g = j >> 5, c = j & 31;
    const float* w = (g == 0) ? w0 : w1;
    dst[i] = (_Float16)w[(size_t)order * K * 32 + (size_t)k * 32 + c];
}
__global__ void k_linpad(const float* __restrict__ w, _Float16* __restrict__ dst) {
    int i = blockIdx.x * 256 + threadIdx.x;          // 32 x 16, zero-padded cols 12..15
    if (i >= 32 * 16) return;
    int k = i >> 4, c = i & 15;
    dst[i] = (c < OUT_) ? (_Float16)w[k * OUT_ + c] : (_Float16)0.f;
}
// Repack row-major f16 [K x Ncols] into WMMA B-fragment order:
//   Bp[((nt*(K/32)+kt)*32 + lane)*16 + e] = Brm[(kt*32 + hi*16 + e)*Ncols + nt*16 + (lane&15)]
__global__ void k_pack(const _Float16* __restrict__ Brm, _Float16* __restrict__ Bp,
                       int K, int Ncols) {
    int i = blockIdx.x * 256 + threadIdx.x;
    if (i >= K * Ncols) return;
    int e    = i & 15;
    int lane = (i >> 4) & 31;
    int ft   = i >> 9;
    int nKt  = K >> 5;
    int kt   = ft % nKt;
    int nt   = ft / nKt;
    int hi   = lane >> 4;
    int col  = (nt << 4) + (lane & 15);
    int k    = (kt << 5) + hi * 16 + e;
    Bp[i] = Brm[(size_t)k * Ncols + col];
}
__global__ void k_biascomb(const float* xz, const float* hz, const float* xr,
                           const float* hr, const float* xh, const float* hh,
                           float* __restrict__ d) {
    int i = threadIdx.x;                             // 96: (x_b + h_b) per gate col
    if (i >= 96) return;
    int g = i >> 5, c = i & 31;
    const float* xb = (g == 0) ? xz : (g == 1) ? xr : xh;
    const float* hb = (g == 0) ? hz : (g == 1) ? hr : hh;
    d[i] = xb[c] + hb[c];
}

// ======================= graph normalization ================================
__global__ void k_deg(const int* __restrict__ src, float* __restrict__ deg) {
    int e = blockIdx.x * 256 + threadIdx.x;
    if (e < E_) atomicAdd(&deg[src[e]], 1.0f);
}
__global__ void k_wneg(const int* __restrict__ src, const int* __restrict__ dst,
                       const float* __restrict__ deg, float* __restrict__ wneg) {
    int e = blockIdx.x * 256 + threadIdx.x;
    if (e >= E_) return;
    float ds = deg[src[e]], dd = deg[dst[e]];
    float is = ds > 0.f ? rsqrtf(fmaxf(ds, 1.f)) : 0.f;
    float id = dd > 0.f ? rsqrtf(fmaxf(dd, 1.f)) : 0.f;
    wneg[e] = -(is * id);
}

// ======================= GAT attention ======================================
__global__ void k_scores(const float* __restrict__ hfeat, const float* __restrict__ as,
                         const float* __restrict__ ad, float* __restrict__ es,
                         float* __restrict__ ed) {
    size_t i = (size_t)blockIdx.x * 256 + threadIdx.x;   // over G*N*HEADS
    if (i >= (size_t)G_ * N_ * HEADS_) return;
    int h = (int)(i & 3);
    size_t gn = i >> 2;
    const float* hp = hfeat + gn * D1_ + h * HID_;
    const float* a0 = as + h * HID_;
    const float* a1 = ad + h * HID_;
    float s = 0.f, d = 0.f;
#pragma unroll
    for (int c = 0; c < HID_; ++c) { float v = hp[c]; s += v * a0[c]; d += v * a1[c]; }
    es[i] = s; ed[i] = d;
}
// softmax denominators (max-shift elided: softmax is shift-invariant, scores O(0.1))
__global__ void k_den(const int* __restrict__ srcv, const int* __restrict__ dstv,
                      const float* __restrict__ es, const float* __restrict__ ed,
                      float* __restrict__ den) {
    size_t i = (size_t)blockIdx.x * 256 + threadIdx.x;   // over G*ESL*HEADS
    if (i >= (size_t)G_ * ESL_ * HEADS_) return;
    int h = (int)(i & 3);
    size_t t = i >> 2;
    int ep = (int)(t % ESL_), g = (int)(t / ESL_);
    int s = ep < E_ ? srcv[ep] : ep - E_;
    int d = ep < E_ ? dstv[ep] : ep - E_;
    float v = es[((size_t)g * N_ + s) * HEADS_ + h] + ed[((size_t)g * N_ + d) * HEADS_ + h];
    v = v > 0.f ? v : 0.2f * v;                          // leaky_relu(0.2)
    atomicAdd(&den[((size_t)g * N_ + d) * HEADS_ + h], expf(v));
}
__global__ void k_aggr(const int* __restrict__ srcv, const int* __restrict__ dstv,
                       const float* __restrict__ es, const float* __restrict__ ed,
                       const float* __restrict__ den, const float* __restrict__ hfeat,
                       float* __restrict__ outb) {
    size_t i = (size_t)blockIdx.x * 256 + threadIdx.x;   // over G*ESL*D1
    if (i >= (size_t)G_ * ESL_ * D1_) return;
    int f = (int)(i & 127);
    size_t t = i >> 7;
    int ep = (int)(t % ESL_), g = (int)(t / ESL_);
    int h = f >> 5;
    int s = ep < E_ ? srcv[ep] : ep - E_;
    int d = ep < E_ ? dstv[ep] : ep - E_;
    float v = es[((size_t)g * N_ + s) * HEADS_ + h] + ed[((size_t)g * N_ + d) * HEADS_ + h];
    v = v > 0.f ? v : 0.2f * v;
    float alpha = expf(v) / den[((size_t)g * N_ + d) * HEADS_ + h];
    atomicAdd(&outb[((size_t)g * N_ + d) * D1_ + f],
              alpha * hfeat[((size_t)g * N_ + s) * D1_ + f]);
}
__global__ void k_biasrelu(float* __restrict__ buf, const float* __restrict__ bias, size_t n) {
    size_t i = (size_t)blockIdx.x * 256 + threadIdx.x;
    if (i >= n) return;
    float v = buf[i] + bias[i & 127];
    buf[i] = v > 0.f ? v : 0.f;
}

// ======================= Chebyshev propagation (SpMM scatter) ===============
__global__ void k_spmm(const int* __restrict__ srcv, const int* __restrict__ dstv,
                       const float* __restrict__ wneg,
                       const float* __restrict__ X, size_t sX,
                       float* __restrict__ P, size_t sP, int F) {
    size_t i = (size_t)blockIdx.x * 256 + threadIdx.x;   // over E*F
    if (i >= (size_t)E_ * F) return;
    int f = (int)(i % F);
    int e = (int)(i / F);
    size_t b = blockIdx.y;
    atomicAdd(&P[b * sP + (size_t)dstv[e] * F + f],
              wneg[e] * X[b * sX + (size_t)srcv[e] * F + f]);
}

// ======================= GRU elementwise ====================================
__global__ void k_gates(const float* __restrict__ XG, const float* __restrict__ HGzr,
                        const float* __restrict__ Hc, float* __restrict__ Z,
                        float* __restrict__ HR) {
    size_t i = (size_t)blockIdx.x * 256 + threadIdx.x;   // B*N*HID
    if (i >= (size_t)B_ * N_ * HID_) return;
    size_t bn = i >> 5; int c = (int)(i & 31);
    float z = XG[bn * 96 + c]      + HGzr[bn * 64 + c];
    float r = XG[bn * 96 + 32 + c] + HGzr[bn * 64 + 32 + c];
    z = 1.f / (1.f + expf(-z));
    r = 1.f / (1.f + expf(-r));
    Z[i]  = z;
    HR[i] = Hc[i] * r;
}
__global__ void k_update(const float* __restrict__ XG, const float* __restrict__ HH,
                         const float* __restrict__ Z, float* __restrict__ Hc) {
    size_t i = (size_t)blockIdx.x * 256 + threadIdx.x;
    if (i >= (size_t)B_ * N_ * HID_) return;
    size_t bn = i >> 5; int c = (int)(i & 31);
    float ht = tanhf(XG[bn * 96 + 64 + c] + HH[i]);
    float z  = Z[i];
    Hc[i] = z * Hc[i] + (1.f - z) * ht;
}

// ======================= host orchestration =================================
extern "C" void kernel_launch(void* const* d_in, const int* in_sizes, int n_in,
                              void* d_out, int out_size, void* d_ws, size_t ws_size,
                              hipStream_t stream)
{
    (void)in_sizes; (void)n_in; (void)out_size; (void)ws_size;
    const float* x    = (const float*)d_in[0];
    const int*   ei   = (const int*)  d_in[1];
    const int*   srcv = ei;
    const int*   dstv = ei + E_;
    const float* g1W  = (const float*)d_in[2];
    const float* g1as = (const float*)d_in[3];
    const float* g1ad = (const float*)d_in[4];
    const float* g1b  = (const float*)d_in[5];
    const float* g2W  = (const float*)d_in[6];
    const float* g2as = (const float*)d_in[7];
    const float* g2ad = (const float*)d_in[8];
    const float* g2b  = (const float*)d_in[9];
    const float* xzW  = (const float*)d_in[10]; const float* xzb = (const float*)d_in[11];
    const float* hzW  = (const float*)d_in[12]; const float* hzb = (const float*)d_in[13];
    const float* xrW  = (const float*)d_in[14]; const float* xrb = (const float*)d_in[15];
    const float* hrW  = (const float*)d_in[16]; const float* hrb = (const float*)d_in[17];
    const float* xhW  = (const float*)d_in[18]; const float* xhb = (const float*)d_in[19];
    const float* hhW  = (const float*)d_in[20]; const float* hhb = (const float*)d_in[21];
    const float* linW = (const float*)d_in[22]; const float* linb = (const float*)d_in[23];
    float* outp = (float*)d_out;

    // ---- workspace carve (≈ 300 MB) ----
    char* ws = (char*)d_ws; size_t off = 0;
    auto alloc = [&](size_t bytes) {
        char* p = ws + off; off = (off + bytes + 255) & ~(size_t)255; return p;
    };
    float* big0 = (float*)alloc((size_t)G_ * N_ * D1_ * 4);      // h = x@W (pre-attention)
    float* big1 = (float*)alloc((size_t)G_ * N_ * D1_ * 4);      // h1 / h2 (GRU sequence)
    float* esb  = (float*)alloc((size_t)G_ * N_ * HEADS_ * 4);
    float* edb  = (float*)alloc((size_t)G_ * N_ * HEADS_ * 4);
    float* den  = (float*)alloc((size_t)G_ * N_ * HEADS_ * 4);
    float* PX   = (float*)alloc((size_t)B_ * N_ * D1_ * 4);
    float* XG   = (float*)alloc((size_t)B_ * N_ * 96 * 4);
    float* Hc   = (float*)alloc((size_t)B_ * N_ * HID_ * 4);
    float* PH   = (float*)alloc((size_t)B_ * N_ * HID_ * 4);
    float* HGzr = (float*)alloc((size_t)B_ * N_ * 64 * 4);
    float* Zb   = (float*)alloc((size_t)B_ * N_ * HID_ * 4);
    float* HR   = (float*)alloc((size_t)B_ * N_ * HID_ * 4);
    float* PHR  = (float*)alloc((size_t)B_ * N_ * HID_ * 4);
    float* HH   = (float*)alloc((size_t)B_ * N_ * HID_ * 4);
    float* wneg = (float*)alloc((size_t)E_ * 4);
    float* deg  = (float*)alloc((size_t)N_ * 4);
    float* bx96 = (float*)alloc(96 * 4);
    // row-major f16 staging
    _Float16* w1h  = (_Float16*)alloc((size_t)FIN_ * D1_ * 2);
    _Float16* w2h  = (_Float16*)alloc((size_t)D1_ * D1_ * 2);
    _Float16* wx0  = (_Float16*)alloc((size_t)D1_ * 96 * 2);
    _Float16* wx1  = (_Float16*)alloc((size_t)D1_ * 96 * 2);
    _Float16* wh0  = (_Float16*)alloc((size_t)32 * 64 * 2);
    _Float16* wh1  = (_Float16*)alloc((size_t)32 * 64 * 2);
    _Float16* whh0 = (_Float16*)alloc((size_t)32 * 32 * 2);
    _Float16* whh1 = (_Float16*)alloc((size_t)32 * 32 * 2);
    _Float16* linh = (_Float16*)alloc((size_t)32 * 16 * 2);
    // WMMA-fragment-packed weights
    _Float16* w1p  = (_Float16*)alloc((size_t)FIN_ * D1_ * 2);
    _Float16* w2p  = (_Float16*)alloc((size_t)D1_ * D1_ * 2);
    _Float16* wx0p = (_Float16*)alloc((size_t)D1_ * 96 * 2);
    _Float16* wx1p = (_Float16*)alloc((size_t)D1_ * 96 * 2);
    _Float16* wh0p = (_Float16*)alloc((size_t)32 * 64 * 2);
    _Float16* wh1p = (_Float16*)alloc((size_t)32 * 64 * 2);
    _Float16* whh0p= (_Float16*)alloc((size_t)32 * 32 * 2);
    _Float16* whh1p= (_Float16*)alloc((size_t)32 * 32 * 2);
    _Float16* linp = (_Float16*)alloc((size_t)32 * 16 * 2);

    auto nb = [](long long n) { return (unsigned)((n + 255) / 256); };

    // ---- weight prep: convert/concat -> repack to fragment order ----
    k_f2h<<<nb(FIN_ * D1_), 256, 0, stream>>>(g1W, w1h, FIN_ * D1_);
    k_f2h<<<nb(D1_ * D1_), 256, 0, stream>>>(g2W, w2h, D1_ * D1_);
    k_cat3<<<nb(D1_ * 96), 256, 0, stream>>>(xzW, xrW, xhW, wx0, D1_, 0);
    k_cat3<<<nb(D1_ * 96), 256, 0, stream>>>(xzW, xrW, xhW, wx1, D1_, 1);
    k_cat2<<<nb(32 * 64), 256, 0, stream>>>(hzW, hrW, wh0, 32, 0);
    k_cat2<<<nb(32 * 64), 256, 0, stream>>>(hzW, hrW, wh1, 32, 1);
    k_f2h<<<nb(1024), 256, 0, stream>>>(hhW, whh0, 1024);
    k_f2h<<<nb(1024), 256, 0, stream>>>(hhW + 1024, whh1, 1024);
    k_linpad<<<nb(512), 256, 0, stream>>>(linW, linh);
    k_pack<<<nb(FIN_ * D1_), 256, 0, stream>>>(w1h, w1p, FIN_, D1_);
    k_pack<<<nb(D1_ * D1_), 256, 0, stream>>>(w2h, w2p, D1_, D1_);
    k_pack<<<nb(D1_ * 96), 256, 0, stream>>>(wx0, wx0p, D1_, 96);
    k_pack<<<nb(D1_ * 96), 256, 0, stream>>>(wx1, wx1p, D1_, 96);
    k_pack<<<nb(32 * 64), 256, 0, stream>>>(wh0, wh0p, 32, 64);
    k_pack<<<nb(32 * 64), 256, 0, stream>>>(wh1, wh1p, 32, 64);
    k_pack<<<nb(32 * 32), 256, 0, stream>>>(whh0, whh0p, 32, 32);
    k_pack<<<nb(32 * 32), 256, 0, stream>>>(whh1, whh1p, 32, 32);
    k_pack<<<nb(32 * 16), 256, 0, stream>>>(linh, linp, 32, 16);
    k_biascomb<<<1, 96, 0, stream>>>(xzb, hzb, xrb, hrb, xhb, hhb, bx96);
    hipMemsetAsync(deg, 0, (size_t)N_ * 4, stream);
    k_deg<<<nb(E_), 256, 0, stream>>>(srcv, deg);
    k_wneg<<<nb(E_), 256, 0, stream>>>(srcv, dstv, deg, wneg);

    const int M48 = G_ * N_;                                    // 240000

    // ---- GAT layer 1: big0 = x@W1; attention -> big1; +b, relu ----
    {
        int tiles = (M48 / 16) * (D1_ / 16);
        k_gemm<<<dim3((tiles + 7) / 8, 1), 256, 0, stream>>>(
            x, 0, FIN_, FIN_, nullptr, 0, 0, 0, w1p, nullptr,
            nullptr, 0, big0, 0, D1_, M48, D1_, D1_);
    }
    k_scores<<<nb((long long)G_ * N_ * HEADS_), 256, 0, stream>>>(big0, g1as, g1ad, esb, edb);
    hipMemsetAsync(den, 0, (size_t)G_ * N_ * HEADS_ * 4, stream);
    hipMemsetAsync(big1, 0, (size_t)G_ * N_ * D1_ * 4, stream);
    k_den<<<nb((long long)G_ * ESL_ * HEADS_), 256, 0, stream>>>(srcv, dstv, esb, edb, den);
    k_aggr<<<nb((long long)G_ * ESL_ * D1_), 256, 0, stream>>>(srcv, dstv, esb, edb, den, big0, big1);
    k_biasrelu<<<nb((long long)G_ * N_ * D1_), 256, 0, stream>>>(big1, g1b, (size_t)G_ * N_ * D1_);

    // ---- GAT layer 2: big0 = big1@W2; attention -> big1; +b, relu ----
    {
        int tiles = (M48 / 16) * (D1_ / 16);
        k_gemm<<<dim3((tiles + 7) / 8, 1), 256, 0, stream>>>(
            big1, 0, D1_, D1_, nullptr, 0, 0, 0, w2p, nullptr,
            nullptr, 0, big0, 0, D1_, M48, D1_, D1_);
    }
    k_scores<<<nb((long long)G_ * N_ * HEADS_), 256, 0, stream>>>(big0, g2as, g2ad, esb, edb);
    hipMemsetAsync(den, 0, (size_t)G_ * N_ * HEADS_ * 4, stream);
    hipMemsetAsync(big1, 0, (size_t)G_ * N_ * D1_ * 4, stream);
    k_den<<<nb((long long)G_ * ESL_ * HEADS_), 256, 0, stream>>>(srcv, dstv, esb, edb, den);
    k_aggr<<<nb((long long)G_ * ESL_ * D1_), 256, 0, stream>>>(srcv, dstv, esb, edb, den, big0, big1);
    k_biasrelu<<<nb((long long)G_ * N_ * D1_), 256, 0, stream>>>(big1, g2b, (size_t)G_ * N_ * D1_);

    // ---- GConvGRU (K=2 ChebConv), 12 steps; seq[t,b] = big1[(b*T+t)] ----
    hipMemsetAsync(Hc, 0, (size_t)B_ * N_ * HID_ * 4, stream);
    const int mT5k = (N_ + 15) / 16;                            // 313 M-tiles
    const size_t sXt = (size_t)T_ * N_ * D1_;                   // batch stride in seq
    const size_t sH  = (size_t)N_ * HID_;
    for (int t = 0; t < T_; ++t) {
        const float* Xt = big1 + (size_t)t * N_ * D1_;
        // PX = prop(Xt)
        hipMemsetAsync(PX, 0, (size_t)B_ * N_ * D1_ * 4, stream);
        k_spmm<<<dim3(nb((long long)E_ * D1_), B_), 256, 0, stream>>>(
            srcv, dstv, wneg, Xt, sXt, PX, (size_t)N_ * D1_, D1_);
        // XG = Xt@Wx0cat + PX@Wx1cat + (xb+hb)   -> [B,N,96] (z|r|h)
        {
            int tiles = mT5k * (96 / 16);
            k_gemm<<<dim3((tiles + 7) / 8, B_), 256, 0, stream>>>(
                Xt, sXt, D1_, D1_, PX, (size_t)N_ * D1_, D1_, D1_,
                wx0p, wx1p, bx96, 0, XG, (size_t)N_ * 96, 96, N_, 96, 96);
        }
        // PH = prop(Hc); HGzr = Hc@Wh0 + PH@Wh1 -> [B,N,64] (z|r)
        hipMemsetAsync(PH, 0, (size_t)B_ * N_ * HID_ * 4, stream);
        k_spmm<<<dim3(nb((long long)E_ * HID_), B_), 256, 0, stream>>>(
            srcv, dstv, wneg, Hc, sH, PH, sH, HID_);
        {
            int tiles = mT5k * (64 / 16);
            k_gemm<<<dim3((tiles + 7) / 8, B_), 256, 0, stream>>>(
                Hc, sH, HID_, HID_, PH, sH, HID_, HID_,
                wh0p, wh1p, nullptr, 0, HGzr, (size_t)N_ * 64, 64, N_, 64, 64);
        }
        // Z, R gates; HR = Hc*R
        k_gates<<<nb((long long)B_ * N_ * HID_), 256, 0, stream>>>(XG, HGzr, Hc, Zb, HR);
        // PHR = prop(HR); HH = HR@hh0 + PHR@hh1
        hipMemsetAsync(PHR, 0, (size_t)B_ * N_ * HID_ * 4, stream);
        k_spmm<<<dim3(nb((long long)E_ * HID_), B_), 256, 0, stream>>>(
            srcv, dstv, wneg, HR, sH, PHR, sH, HID_);
        {
            int tiles = mT5k * (32 / 16);
            k_gemm<<<dim3((tiles + 7) / 8, B_), 256, 0, stream>>>(
                HR, sH, HID_, HID_, PHR, sH, HID_, HID_,
                whh0p, whh1p, nullptr, 0, HH, sH, HID_, N_, HID_, HID_);
        }
        // Hc = Z*Hc + (1-Z)*tanh(XG_h + HH)
        k_update<<<nb((long long)B_ * N_ * HID_), 256, 0, stream>>>(XG, HH, Zb, Hc);
    }

    // ---- final linear: out = Hc @ lin_W + lin_b  ([20000,32]@[32,12]) ----
    {
        int M = B_ * N_;
        int tiles = (M / 16) * 1;                               // N padded to one 16-tile
        k_gemm<<<dim3((tiles + 7) / 8, 1), 256, 0, stream>>>(
            Hc, 0, HID_, HID_, nullptr, 0, 0, 0, linp, nullptr,
            linb, 0, outp, 0, OUT_, M, 16, OUT_);
    }
}
